// SubqueryFeatureExtractor_77051713290508
// MI455X (gfx1250) — matrix operand order repaired and checked by reading, not compiled, not run on hardware
//
#include <hip/hip_runtime.h>
#include <hip/hip_bf16.h>

typedef __attribute__((ext_vector_type(16))) _Float16 v16h;
typedef __attribute__((ext_vector_type(8)))  _Float16 v8h;
typedef __attribute__((ext_vector_type(8)))  float    v8f;

#define Bn 4
#define Nq 500
#define Cc 256
#define Hh 8
#define DH 32
#define Pp 32
#define Gg 4
#define CG 64
#define Mr 2000   // B*N rows
#define Kc 256

// ---------------------------------------------------------------- converts
__global__ void f32_to_f16_kernel(const float* __restrict__ src,
                                  _Float16* __restrict__ dst, int n) {
    int i = blockIdx.x * blockDim.x + threadIdx.x;
    if (i < n) dst[i] = (_Float16)src[i];
}

// ------------------------------------------------- position embedding + add
__global__ __launch_bounds__(256)
void pe_add_kernel(const float* __restrict__ qcontent,
                   const float* __restrict__ qbox,
                   float* __restrict__ x, _Float16* __restrict__ xh) {
    int row = blockIdx.x;          // b*N + n
    int c   = threadIdx.x;         // 0..255
    const float* bx = &qbox[row * 4];
    int t = c >> 6;                // which box term
    int r = c & 63;
    int i = r >> 1;
    float term  = (t < 2) ? 1000.0f : 1.0f;
    float token = bx[t] / term;
    float freq  = __powf(10000.0f, (float)i * (2.0f / 64.0f));
    float val   = token / freq;
    float pe    = (r & 1) ? __cosf(val) : __sinf(val);
    float xv    = qcontent[row * Cc + c] + pe;
    x[row * Cc + c]  = xv;
    xh[row * Cc + c] = (_Float16)xv;
}

// ------------------------------------------------------------- IoF log bias
__device__ __forceinline__ void decode_box4(const float* p, float* box) {
    float sc = exp2f(p[2]);
    float w  = sc * exp2f(-0.5f * p[3]);
    float h  = sc * exp2f( 0.5f * p[3]);
    box[0] = p[0] - 0.5f * w; box[1] = p[1] - 0.5f * h;
    box[2] = p[0] + 0.5f * w; box[3] = p[1] + 0.5f * h;
}

__global__ void iof_kernel(const float* __restrict__ qbox,
                           float* __restrict__ iofl) {
    int idx = blockIdx.x * 256 + threadIdx.x;
    if (idx >= Bn * Nq * Nq) return;
    int m = idx % Nq;
    int n = (idx / Nq) % Nq;
    int b = idx / (Nq * Nq);
    float bnb[4], bmb[4];
    decode_box4(&qbox[(b * Nq + n) * 4], bnb);
    decode_box4(&qbox[(b * Nq + m) * 4], bmb);
    float area = fmaxf(bnb[2] - bnb[0], 0.f) * fmaxf(bnb[3] - bnb[1], 0.f);
    float ltx = fmaxf(bnb[0], bmb[0]), lty = fmaxf(bnb[1], bmb[1]);
    float rbx = fminf(bnb[2], bmb[2]), rby = fminf(bnb[3], bmb[3]);
    float inter = fmaxf(rbx - ltx, 0.f) * fmaxf(rby - lty, 0.f);
    float iof = inter / fmaxf(area, 1e-7f);
    iofl[idx] = __logf(iof + 1e-7f);
}

// ------------------------------------------------------ WMMA GEMM Y = X*W^T
// X: Mx256 f16 row-major, W: Ncx256 f16 row-major (out,in), Y: MxNc f32.
// Both operands are K-major => per-lane fragments are contiguous in memory:
// load them straight from global (b128s), no LDS staging, no divergent guards.
// Block tile 64(M) x 64(N); 8 wave32 waves; each wave one 16x32 tile (2 WMMA).
__global__ __launch_bounds__(256)
void wmma_gemm_xwt(const _Float16* __restrict__ Xh,
                   const _Float16* __restrict__ Wh,
                   const float* __restrict__ bias,
                   float* __restrict__ Y,
                   int M, int Nc) {
    int tid  = threadIdx.x;
    int wid  = tid >> 5;
    int lane = tid & 31;
    int halfsel = lane >> 4;
    int l16  = lane & 15;
    int bm = blockIdx.x * 64;
    int bn = blockIdx.y * 64;
    int wm = (wid >> 1) * 16;     // 0,16,32,48
    int wn = (wid & 1) * 32;      // 0,32

    // A fragment source: row (bm+wm+l16), K chunk ka..ka+15 (contiguous)
    int arow = bm + wm + l16;
    arow = (arow < M) ? arow : (M - 1);          // clamp: garbage rows never stored
    const _Float16* aptr  = Xh + (size_t)arow * Kc + halfsel * 16;
    // B fragments: col rows of W, K chunk kb..kb+15 (contiguous 32B)
    const _Float16* bptr0 = Wh + (size_t)(bn + wn + l16) * Kc + halfsel * 16;
    const _Float16* bptr1 = Wh + (size_t)(bn + wn + 16 + l16) * Kc + halfsel * 16;

    v8f acc0 = {0.f, 0.f, 0.f, 0.f, 0.f, 0.f, 0.f, 0.f};
    v8f acc1 = {0.f, 0.f, 0.f, 0.f, 0.f, 0.f, 0.f, 0.f};

    #pragma unroll
    for (int k0 = 0; k0 < Kc; k0 += 32) {
        v8h alo = *(const v8h*)(aptr + k0);          // 16B aligned
        v8h ahi = *(const v8h*)(aptr + k0 + 8);      // 16B aligned
        v16h a  = __builtin_shufflevector(alo, ahi,
                      0, 1, 2, 3, 4, 5, 6, 7, 8, 9, 10, 11, 12, 13, 14, 15);
        v16h b0 = *(const v16h*)(bptr0 + k0);        // 32B aligned
        v16h b1 = *(const v16h*)(bptr1 + k0);        // 32B aligned
        acc0 = __builtin_amdgcn_wmma_f32_16x16x32_f16(
                   false, a, false, b0, (short)0, acc0, false, false);
        acc1 = __builtin_amdgcn_wmma_f32_16x16x32_f16(
                   false, a, false, b1, (short)0, acc1, false, false);
    }
    // C/D layout: VGPR r -> row M = r + halfsel*8, col = l16
    int col0 = bn + wn + l16;
    int col1 = col0 + 16;
    float bia0 = bias[col0];
    float bia1 = bias[col1];
    #pragma unroll
    for (int r = 0; r < 8; ++r) {
        int row = bm + wm + halfsel * 8 + r;
        if (row < M) {
            Y[(size_t)row * Nc + col0] = acc0[r] + bia0;
            Y[(size_t)row * Nc + col1] = acc1[r] + bia1;
        }
    }
}

// ------------------------------------------------------------- attention
__global__ __launch_bounds__(256)
void attn_kernel(const float* __restrict__ q, const float* __restrict__ k,
                 const float* __restrict__ v, const float* __restrict__ iofl,
                 const float* __restrict__ tau, float* __restrict__ ctx) {
    int n = blockIdx.x, h = blockIdx.y, b = blockIdx.z;
    __shared__ float sQ[32];
    __shared__ float sS[512];
    __shared__ float red[256];
    __shared__ float redc[8][33];
    int tid = threadIdx.x;
    const float scale = 0.1767766952966369f;   // 1/sqrt(32)
    int qbase = (b * Nq + n) * Cc + h * DH;
    if (tid < 32) sQ[tid] = q[qbase + tid];
    __syncthreads();
    float th = tau[h];
    float lmax = -1e30f;
    for (int m = tid; m < Nq; m += 256) {
        const float* kp = &k[(b * Nq + m) * Cc + h * DH];
        float s = 0.f;
        #pragma unroll
        for (int d = 0; d < DH; ++d) s += sQ[d] * kp[d];
        s = s * scale + iofl[(b * Nq + n) * Nq + m] * th;
        sS[m] = s;
        lmax = fmaxf(lmax, s);
    }
    red[tid] = lmax; __syncthreads();
    for (int o = 128; o > 0; o >>= 1) {
        if (tid < o) red[tid] = fmaxf(red[tid], red[tid + o]);
        __syncthreads();
    }
    float mx = red[0]; __syncthreads();
    float lsum = 0.f;
    for (int m = tid; m < Nq; m += 256) {
        float e = __expf(sS[m] - mx);
        sS[m] = e;
        lsum += e;
    }
    red[tid] = lsum; __syncthreads();
    for (int o = 128; o > 0; o >>= 1) {
        if (tid < o) red[tid] += red[tid + o];
        __syncthreads();
    }
    float inv = 1.0f / red[0];
    // ctx[d] = sum_m p_m * v[m,d]
    int d = tid & 31, i = tid >> 5;
    float part = 0.f;
    for (int m = i; m < Nq; m += 8)
        part += sS[m] * v[(b * Nq + m) * Cc + h * DH + d];
    redc[i][d] = part; __syncthreads();
    if (i == 0) {
        float s = 0.f;
        #pragma unroll
        for (int j = 0; j < 8; ++j) s += redc[j][d];
        ctx[qbase + d] = s * inv;
    }
}

// ------------------------------------------------------------ layernorm
__global__ __launch_bounds__(256)
void ln_kernel(const float* __restrict__ x, const float* __restrict__ cwo,
               const float* __restrict__ g, const float* __restrict__ bta,
               float* __restrict__ qc_out, _Float16* __restrict__ qc_h) {
    int row = blockIdx.x;
    int c   = threadIdx.x;
    __shared__ float red[256];
    float val = x[row * Cc + c] + cwo[row * Cc + c];   // bo folded into GEMM bias
    red[c] = val; __syncthreads();
    for (int o = 128; o > 0; o >>= 1) {
        if (c < o) red[c] += red[c + o];
        __syncthreads();
    }
    float mu = red[0] * (1.0f / Cc); __syncthreads();
    float dv = val - mu;
    red[c] = dv * dv; __syncthreads();
    for (int o = 128; o > 0; o >>= 1) {
        if (c < o) red[c] += red[c + o];
        __syncthreads();
    }
    float var = red[0] * (1.0f / Cc);
    float out = dv * rsqrtf(var + 1e-5f) * g[c] + bta[c];
    qc_out[row * Cc + c] = out;
    qc_h[row * Cc + c]   = (_Float16)out;
}

// ------------------------------------------- bilinear pyramid sampler
__global__ __launch_bounds__(64)
void sample_kernel(const float* __restrict__ f0, const float* __restrict__ f1,
                   const float* __restrict__ f2, const float* __restrict__ f3,
                   const float* __restrict__ qbox,
                   const float* __restrict__ offset,    // (2000,256) = (B,N,P,G,2)
                   const float* __restrict__ sclogit,   // (2000,128) = (B,N,P,G)
                   float* __restrict__ sfeat,           // (B,N,G,P,64)
                   float* __restrict__ sxy_out,         // (B,N,P,G,2)
                   float* __restrict__ sz_out) {        // (B,N,P,G)
    int idx = blockIdx.x;              // ((b*N+n)*P + p)*G + g
    int g  = idx & 3;
    int p  = (idx >> 2) & 31;
    int bn = idx >> 7;                 // b*N + n
    int b  = bn / Nq;
    int c  = threadIdx.x;              // channel within group, 0..63

    const float* qb = &qbox[bn * 4];
    float cx = qb[0], cy = qb[1], z = qb[2], rr = qb[3];
    float sc = exp2f(z);
    float wx = sc * exp2f(-0.5f * rr);
    float wy = sc * exp2f( 0.5f * rr);
    float ox = offset[idx * 2 + 0], oy = offset[idx * 2 + 1];
    float sx = cx + ox * wx, sy = cy + oy * wy;
    float sz = sclogit[idx] + z;
    float lvl = sz - 3.0f;
    float zw[4]; float zsum = 0.f;
    #pragma unroll
    for (int l = 0; l < 4; ++l) {
        float d = lvl - (float)l;
        zw[l] = __expf(-d * d * 0.5f);   // TAU_Z = 2
        zsum += zw[l];
    }
    float zinv = 1.0f / zsum;

    const float* feats[4] = {f0, f1, f2, f3};
    const int   HH[4]  = {128, 64, 32, 16};
    const float STR[4] = {8.f, 16.f, 32.f, 64.f};

    // warm L2 for this lane's channel planes (global_prefetch_b8)
    #pragma unroll
    for (int l = 0; l < 4; ++l)
        __builtin_prefetch(feats[l] + (size_t)(b * Cc + g * CG + c) * HH[l] * HH[l], 0, 1);

    float acc = 0.f;
    #pragma unroll
    for (int l = 0; l < 4; ++l) {
        int Hl = HH[l], Wl = HH[l];
        float px = sx / STR[l] - 0.5f;
        float py = sy / STR[l] - 0.5f;
        float x0f = floorf(px), y0f = floorf(py);
        float fx = px - x0f, fy = py - y0f;
        int x0 = (int)x0f, y0 = (int)y0f;
        bool xi0 = (x0 >= 0) && (x0 <= Wl - 1);
        bool xi1 = (x0 + 1 >= 0) && (x0 + 1 <= Wl - 1);
        bool yi0 = (y0 >= 0) && (y0 <= Hl - 1);
        bool yi1 = (y0 + 1 >= 0) && (y0 + 1 <= Hl - 1);
        int x0c = min(max(x0, 0), Wl - 1);
        int x1c = min(max(x0 + 1, 0), Wl - 1);
        int y0c = min(max(y0, 0), Hl - 1);
        int y1c = min(max(y0 + 1, 0), Hl - 1);
        float w00 = (1.f - fx) * (1.f - fy) * ((xi0 && yi0) ? 1.f : 0.f);
        float w01 = fx * (1.f - fy)         * ((xi1 && yi0) ? 1.f : 0.f);
        float w10 = (1.f - fx) * fy         * ((xi0 && yi1) ? 1.f : 0.f);
        float w11 = fx * fy                 * ((xi1 && yi1) ? 1.f : 0.f);
        const float* base = feats[l] + (size_t)(b * Cc + g * CG + c) * Hl * Wl;
        float bil = base[y0c * Wl + x0c] * w00 + base[y0c * Wl + x1c] * w01 +
                    base[y1c * Wl + x0c] * w10 + base[y1c * Wl + x1c] * w11;
        acc += bil * zw[l] * zinv;
    }
    sfeat[(size_t)((bn * Gg + g) * Pp + p) * CG + c] = acc;
    if (c == 0) {
        sxy_out[idx * 2 + 0] = sx;
        sxy_out[idx * 2 + 1] = sy;
        sz_out[idx] = sz;
    }
}

// ================================================================== host
extern "C" void kernel_launch(void* const* d_in, const int* in_sizes, int n_in,
                              void* d_out, int out_size, void* d_ws, size_t ws_size,
                              hipStream_t stream) {
    const float* feat0 = (const float*)d_in[0];
    const float* feat1 = (const float*)d_in[1];
    const float* feat2 = (const float*)d_in[2];
    const float* feat3 = (const float*)d_in[3];
    const float* qcont = (const float*)d_in[4];
    const float* qbox  = (const float*)d_in[5];
    const float* Wq = (const float*)d_in[10]; const float* bq = (const float*)d_in[11];
    const float* Wk = (const float*)d_in[12]; const float* bk = (const float*)d_in[13];
    const float* Wv = (const float*)d_in[14]; const float* bv = (const float*)d_in[15];
    const float* Wo = (const float*)d_in[16]; const float* bo = (const float*)d_in[17];
    const float* lng = (const float*)d_in[18]; const float* lnb = (const float*)d_in[19];
    const float* offW = (const float*)d_in[20]; const float* offb = (const float*)d_in[21];
    const float* scW  = (const float*)d_in[22]; const float* scb  = (const float*)d_in[23];
    const float* tau  = (const float*)d_in[24];

    // ---- d_out layout (element offsets, reference tuple order)
    float* out = (float*)d_out;
    const size_t OFF_FEATS  = 0;                     // (B,N,G,P,64)  16,384,000
    const size_t OFF_SQXY   = 16384000;              //   512,000
    const size_t OFF_SQZ    = OFF_SQXY  +   512000;  // 1,024,000
    const size_t OFF_SQV    = OFF_SQZ   +  1024000;  // 16,384,000
    const size_t OFF_QC     = OFF_SQV   + 16384000;  //   512,000
    const size_t OFF_SXY    = OFF_QC    +   512000;  //   512,000
    const size_t OFF_OFFSET = OFF_SXY   +   512000;  //   512,000
    const size_t OFF_SZ     = OFF_OFFSET+   512000;  //   256,000
    const size_t OFF_SCL    = OFF_SZ    +   256000;  //   256,000

    // ---- workspace carve
    char* wsp = (char*)d_ws;
    auto carve = [&](size_t bytes) {
        void* pp = (void*)wsp;
        wsp += (bytes + 255) & ~(size_t)255;
        return pp;
    };
    float*    x_f   = (float*)   carve((size_t)Mr * Cc * 4);
    _Float16* x_h   = (_Float16*)carve((size_t)Mr * Cc * 2);
    _Float16* wq_h  = (_Float16*)carve(256 * 256 * 2);
    _Float16* wk_h  = (_Float16*)carve(256 * 256 * 2);
    _Float16* wv_h  = (_Float16*)carve(256 * 256 * 2);
    _Float16* wo_h  = (_Float16*)carve(256 * 256 * 2);
    _Float16* offw_h= (_Float16*)carve(256 * 256 * 2);
    _Float16* scw_h = (_Float16*)carve(128 * 256 * 2);
    float*    q_f   = (float*)   carve((size_t)Mr * Cc * 4);
    float*    k_f   = (float*)   carve((size_t)Mr * Cc * 4);
    float*    v_f   = (float*)   carve((size_t)Mr * Cc * 4);
    float*    iofl  = (float*)   carve((size_t)Bn * Nq * Nq * 4);
    float*    ctx_f = (float*)   carve((size_t)Mr * Cc * 4);
    _Float16* ctx_h = (_Float16*)carve((size_t)Mr * Cc * 2);
    float*    cwo_f = (float*)   carve((size_t)Mr * Cc * 4);
    _Float16* qc_h  = (_Float16*)carve((size_t)Mr * Cc * 2);

    auto cvt = [&](const float* s, _Float16* d, int n) {
        f32_to_f16_kernel<<<(n + 255) / 256, 256, 0, stream>>>(s, d, n);
    };
    // 1) weight converts
    cvt(Wq, wq_h, 65536);   cvt(Wk, wk_h, 65536);
    cvt(Wv, wv_h, 65536);   cvt(Wo, wo_h, 65536);
    cvt(offW, offw_h, 65536); cvt(scW, scw_h, 32768);

    // 2) positional embedding + residual input
    pe_add_kernel<<<Mr, 256, 0, stream>>>(qcont, qbox, x_f, x_h);

    // 3) IoF log bias
    iof_kernel<<<(Bn * Nq * Nq + 255) / 256, 256, 0, stream>>>(qbox, iofl);

    // 4) QKV projections via WMMA
    auto gemm = [&](const _Float16* X, const _Float16* Wm, const float* bias,
                    float* Y, int Nc) {
        dim3 grid((Mr + 63) / 64, Nc / 64);
        wmma_gemm_xwt<<<grid, 256, 0, stream>>>(X, Wm, bias, Y, Mr, Nc);
    };
    gemm(x_h, wq_h, bq, q_f, 256);
    gemm(x_h, wk_h, bk, k_f, 256);
    gemm(x_h, wv_h, bv, v_f, 256);

    // 5) attention with IoF bias
    attn_kernel<<<dim3(Nq, Hh, Bn), 256, 0, stream>>>(q_f, k_f, v_f, iofl, tau, ctx_f);

    // 6) ctx -> f16, Wo projection (bo folded into GEMM bias)
    cvt(ctx_f, ctx_h, Mr * Cc);
    gemm(ctx_h, wo_h, bo, cwo_f, 256);

    // 7) residual + layernorm -> qc (d_out) + f16 copy
    ln_kernel<<<Mr, 256, 0, stream>>>(x_f, cwo_f, lng, lnb, out + OFF_QC, qc_h);

    // 8) offset / scale heads (written straight into d_out)
    gemm(qc_h, offw_h, offb, out + OFF_OFFSET, 256);
    gemm(qc_h, scw_h,  scb,  out + OFF_SCL,   128);

    // 9) pyramid bilinear sampling
    sample_kernel<<<Bn * Nq * Pp * Gg, 64, 0, stream>>>(
        feat0, feat1, feat2, feat3, qbox,
        out + OFF_OFFSET, out + OFF_SCL,
        out + OFF_FEATS, out + OFF_SXY, out + OFF_SZ);

    // 10) pass-through outputs
    hipMemcpyAsync(out + OFF_SQXY, d_in[6],  512000 * sizeof(float),
                   hipMemcpyDeviceToDevice, stream);
    hipMemcpyAsync(out + OFF_SQZ,  d_in[7], 1024000 * sizeof(float),
                   hipMemcpyDeviceToDevice, stream);
    hipMemcpyAsync(out + OFF_SQV,  d_in[8], 16384000 * sizeof(float),
                   hipMemcpyDeviceToDevice, stream);

    (void)in_sizes; (void)n_in; (void)out_size; (void)ws_size;
}
